// EquivariantGenerativeModel_3393024164023
// MI455X (gfx1250) — matrix operand order repaired and checked by reading, not compiled, not run on hardware
//
#include <hip/hip_runtime.h>
#include <math.h>
#include <stdint.h>
#include <stddef.h>

// ---------------------------------------------------------------------------
// Model dimensions (match the JAX reference)
// ---------------------------------------------------------------------------
constexpr int kB   = 32;
constexpr int kN   = 128;
constexpr int kBUF = 64;
constexpr int kC   = 1024;
constexpr int kH   = 16;
constexpr int kL   = 16;
constexpr int kVAE = 16;
constexpr int kHD  = 64;      // kC / kH
constexpr int kMLP = 4096;    // 4*C
constexpr int kS   = 192;     // BUF + N

#define DEVI __device__ __forceinline__

typedef __bf16 bf16_t;
typedef __bf16 v16bf __attribute__((ext_vector_type(16)));
typedef __bf16 v8bf  __attribute__((ext_vector_type(8)));
typedef float  v8f   __attribute__((ext_vector_type(8)));
typedef int    v4i   __attribute__((vector_size(16)));   // 16-byte chunk type

// ---------------------------------------------------------------------------
// CDNA5 async global->LDS copy (ASYNCcnt path), with synchronous fallback
// ---------------------------------------------------------------------------
#if defined(__has_builtin)
#  if __has_builtin(__builtin_amdgcn_global_load_async_to_lds_b128)
#    define ATH_ASYNC_LDS 1
#  endif
#endif

DEVI void async_cp16(const bf16_t* g, bf16_t* l) {
#ifdef ATH_ASYNC_LDS
  __builtin_amdgcn_global_load_async_to_lds_b128(
      (__attribute__((address_space(1))) v4i*)(void*)g,
      (__attribute__((address_space(3))) v4i*)(void*)l,
      0, 0);
#else
  *(v8bf*)l = *(const v8bf*)g;        // synchronous 16B copy fallback
#endif
}
DEVI void async_wait_le3() {
#ifdef ATH_ASYNC_LDS
  asm volatile("s_wait_asynccnt 3" ::: "memory");
#endif
}
DEVI void async_wait_le0() {
#ifdef ATH_ASYNC_LDS
  asm volatile("s_wait_asynccnt 0" ::: "memory");
#endif
}

// ---------------------------------------------------------------------------
// WMMA helpers (v_wmma_f32_16x16x32_bf16, wave32)
// ---------------------------------------------------------------------------
DEVI v8f wmma_bf16(v16bf a, v16bf b, v8f c) {
  return __builtin_amdgcn_wmma_f32_16x16x32_bf16(
      /*neg_a=*/false, a, /*neg_b=*/false, b,
      /*c_mod=*/(short)0, c, /*reuse_a=*/false, /*reuse_b=*/false);
}

// A-operand 16(M)x32(K): lane l -> row = l&15, khalf g = l>>4.
// Elements 0-15 hold K = g*16 .. g*16+15 (contiguous, matches B-frag).
DEVI v16bf load_a_frag(const bf16_t* base, int ld, int kb) {
  const int lane = threadIdx.x & 31;
  const int row  = lane & 15;
  const int g    = lane >> 4;
  const bf16_t* p = base + (size_t)row * ld + kb + g * 16;
  v8bf lo = *(const v8bf*)p;
  v8bf hi = *(const v8bf*)(p + 8);
  v16bf r;
#pragma unroll
  for (int i = 0; i < 8; ++i) { r[i] = lo[i]; r[i + 8] = hi[i]; }
  return r;
}

// B-operand 32(K)x16(N) from a row-major [N,K] matrix (i.e. B = W^T):
// lane l -> col = l&15, kgroup g = l>>4; reads 16 contiguous bf16 at K=g*16.
DEVI v16bf load_b_frag(const bf16_t* base, int ld, int kb) {
  const int lane = threadIdx.x & 31;
  const int col  = lane & 15;
  const int g    = lane >> 4;
  return *(const v16bf*)(base + (size_t)col * ld + kb + g * 16);
}

// ---------------------------------------------------------------------------
// GEMM: out[M,Nn] = A[M,K](bf16) * W[Nn,K]^T(bf16) + bias (+gelu) (+res)
// 256 threads = 8 waves (4 M-tiles x 2 N-tiles), WG tile 128x64, K-step 32.
// Each wave computes a 32x32 output block = 4 WMMAs from 2 A-frags x 2 B-frags
// (register reuse). Double-buffered LDS tiles fed by async global->LDS copies
// (3 async b128 per wave per K-step -> pipelined wait level 3).
// Output rows optionally remapped: orow = (m/oseg)*ostr + ooff + m%oseg
// ---------------------------------------------------------------------------
__global__ __launch_bounds__(256)
void gemm_bf16_kernel(const bf16_t* __restrict__ A, const bf16_t* __restrict__ W,
                      const float* __restrict__ bias, const float* __restrict__ res,
                      float* __restrict__ outF, bf16_t* __restrict__ outB,
                      int M, int Nn, int K, int flags /*1=gelu*/,
                      int oseg, int ooff, int ostr) {
  __shared__ __align__(64) bf16_t lA[2][128 * 32];   // 8KB per buffer
  __shared__ __align__(64) bf16_t lW[2][64 * 32];    // 4KB per buffer

  const int tileM = blockIdx.y * 128;
  const int tileN = blockIdx.x * 64;
  const int tid   = threadIdx.x;
  const int wid   = tid >> 5, lane = tid & 31;
  const int wm    = wid & 3,  wn   = wid >> 2;   // 4x2 wave grid
  const int g     = lane >> 4, nn  = lane & 15;

  // Staging maps: A = 128 rows x 32 elems (2 x 16B per thread),
  //               W = 64 rows x 32 elems (1 x 16B per thread).
  const int ra  = tid >> 1;            // 0..127
  const int cha = (tid & 1) * 16;      // 0 or 16 (elements)
  const int rw  = tid >> 2;            // 0..63
  const int chw = (tid & 3) * 8;       // 0,8,16,24 (elements)

  // Prologue: stage K-tile 0 into buffer 0.
  {
    const bf16_t* arow = A + (size_t)(tileM + ra) * K + cha;
    async_cp16(arow,     &lA[0][ra * 32 + cha]);
    async_cp16(arow + 8, &lA[0][ra * 32 + cha + 8]);
    async_cp16(W + (size_t)(tileN + rw) * K + chw, &lW[0][rw * 32 + chw]);
  }

  v8f zero = {0.f, 0.f, 0.f, 0.f, 0.f, 0.f, 0.f, 0.f};
  v8f acc[2][2];
#pragma unroll
  for (int i = 0; i < 2; ++i)
#pragma unroll
    for (int j = 0; j < 2; ++j) acc[i][j] = zero;

  const int nk = K >> 5;
  for (int kt = 0; kt < nk; ++kt) {
    if (kt + 1 < nk) {
      const int kb2  = (kt + 1) << 5;
      const int bnxt = (kt + 1) & 1;
      const bf16_t* arow = A + (size_t)(tileM + ra) * K + kb2 + cha;
      async_cp16(arow,     &lA[bnxt][ra * 32 + cha]);
      async_cp16(arow + 8, &lA[bnxt][ra * 32 + cha + 8]);
      async_cp16(W + (size_t)(tileN + rw) * K + kb2 + chw, &lW[bnxt][rw * 32 + chw]);
      async_wait_le3();               // current tile's 3 copies have completed
    } else {
      async_wait_le0();
    }
    __syncthreads();

    const int bsel = kt & 1;
    v16bf a0 = load_a_frag(&lA[bsel][(wm * 32) * 32], 32, 0);
    v16bf a1 = load_a_frag(&lA[bsel][(wm * 32 + 16) * 32], 32, 0);
    v16bf b0 = load_b_frag(&lW[bsel][(wn * 32) * 32], 32, 0);
    v16bf b1 = load_b_frag(&lW[bsel][(wn * 32 + 16) * 32], 32, 0);
    acc[0][0] = wmma_bf16(a0, b0, acc[0][0]);
    acc[0][1] = wmma_bf16(a0, b1, acc[0][1]);
    acc[1][0] = wmma_bf16(a1, b0, acc[1][0]);
    acc[1][1] = wmma_bf16(a1, b1, acc[1][1]);
    __syncthreads();                  // protect buffer reuse across waves
  }

  // Epilogue: D layout -> M = am*16 + g*8 + rr, N = p*16 + nn within wave tile
#pragma unroll
  for (int am = 0; am < 2; ++am) {
#pragma unroll
    for (int rr = 0; rr < 8; ++rr) {
      const int m = tileM + wm * 32 + am * 16 + g * 8 + rr;
      const size_t orow = (size_t)(m / oseg) * ostr + ooff + (m % oseg);
#pragma unroll
      for (int p = 0; p < 2; ++p) {
        const int col = tileN + wn * 32 + p * 16 + nn;
        float v = acc[am][p][rr];
        if (bias) v += bias[col];
        if (flags & 1) v = 0.5f * v * (1.0f + erff(v * 0.70710678118654752f));
        const size_t oi = orow * (size_t)Nn + col;
        if (res)  v += res[oi];
        if (outF) outF[oi] = v;
        if (outB) outB[(size_t)m * Nn + col] = (bf16_t)v;
      }
    }
  }
}

// ---------------------------------------------------------------------------
// Attention (flash-style, per-wave 16-query block). Q,K: [B,H,rows,HD] bf16,
// Vt: [B,H,HD,KEYS] bf16 (pre-transposed), out: [B,N,C] bf16.
// grid = (N/16, H, B), block = 32 (one wave).
// ---------------------------------------------------------------------------
template <int KEYS, bool CAUSAL>
__global__ __launch_bounds__(32)
void attn_wmma_kernel(const bf16_t* __restrict__ Q, const bf16_t* __restrict__ Km,
                      const bf16_t* __restrict__ Vt, bf16_t* __restrict__ out,
                      float scale) {
  constexpr int CT = KEYS / 16;
  __shared__ __align__(64) bf16_t P[16 * KEYS];

  const int qt = blockIdx.x, h = blockIdx.y, b = blockIdx.z;
  const int lane = threadIdx.x;
  const int g = lane >> 4, nn = lane & 15;

  const bf16_t* qbase = Q  + (((size_t)b * kH + h) * kN + qt * 16) * kHD;
  const bf16_t* kbase = Km + ((size_t)b * kH + h) * (size_t)KEYS * kHD;
  const bf16_t* vbase = Vt + ((size_t)b * kH + h) * (size_t)kHD * KEYS;

  v8f zero = {0.f, 0.f, 0.f, 0.f, 0.f, 0.f, 0.f, 0.f};
  v8f s[CT];
#pragma unroll
  for (int ct = 0; ct < CT; ++ct) s[ct] = zero;

  // S = Q * K^T  (K stored [key, d] row-major -> contiguous B-frag loads)
#pragma unroll
  for (int kb = 0; kb < kHD; kb += 32) {
    v16bf a = load_a_frag(qbase, kHD, kb);
#pragma unroll
    for (int ct = 0; ct < CT; ++ct) {
      v16bf bf = load_b_frag(kbase + (size_t)ct * 16 * kHD, kHD, kb);
      s[ct] = wmma_bf16(a, bf, s[ct]);
    }
  }

  // Per-row masked softmax: row lives in one VGPR index across 16 lanes of a
  // half-wave -> reduce with width-16 xor shuffles.
#pragma unroll
  for (int rr = 0; rr < 8; ++rr) {
    const int qi = qt * 16 + g * 8 + rr;
    float vals[CT];
    float mx = -3.0e38f;
#pragma unroll
    for (int ct = 0; ct < CT; ++ct) {
      float v = s[ct][rr] * scale;
      if (CAUSAL && (ct * 16 + nn) > qi) v = -3.0e38f;
      vals[ct] = v;
      mx = fmaxf(mx, v);
    }
#pragma unroll
    for (int m = 1; m < 16; m <<= 1) mx = fmaxf(mx, __shfl_xor(mx, m, 16));
    float sum = 0.f;
#pragma unroll
    for (int ct = 0; ct < CT; ++ct) { float e = expf(vals[ct] - mx); vals[ct] = e; sum += e; }
#pragma unroll
    for (int m = 1; m < 16; m <<= 1) sum += __shfl_xor(sum, m, 16);
    const float inv = 1.f / sum;
#pragma unroll
    for (int ct = 0; ct < CT; ++ct)
      P[(g * 8 + rr) * KEYS + ct * 16 + nn] = (bf16_t)(vals[ct] * inv);
  }
  __syncthreads();

  // O = P * V  (Vt stored [d, key] row-major -> contiguous B-frag loads)
#pragma unroll
  for (int dt = 0; dt < kHD / 16; ++dt) {
    v8f o = zero;
#pragma unroll
    for (int kb = 0; kb < KEYS; kb += 32) {
      v16bf a  = load_a_frag(P, KEYS, kb);
      v16bf bf = load_b_frag(vbase + (size_t)dt * 16 * KEYS, KEYS, kb);
      o = wmma_bf16(a, bf, o);
    }
#pragma unroll
    for (int rr = 0; rr < 8; ++rr) {
      const int m = qt * 16 + g * 8 + rr;
      out[((size_t)b * kN + m) * kC + h * kHD + dt * 16 + nn] = (bf16_t)o[rr];
    }
  }
}

// ---------------------------------------------------------------------------
// LayerNorm over C=1024. Input rows remapped via (seg, soff, sstr); outputs
// are compact. outF (f32) and/or outB (bf16) may be null. 1 block per row.
// ---------------------------------------------------------------------------
__global__ __launch_bounds__(256)
void layernorm_kernel(const float* __restrict__ in, const float* __restrict__ w,
                      const float* __restrict__ bb, float* __restrict__ outF,
                      bf16_t* __restrict__ outB, int seg, int soff, int sstr,
                      float eps) {
  const int r = blockIdx.x;
  const int inrow = (r / seg) * sstr + soff + (r % seg);
  const float* x = in + (size_t)inrow * kC;

  float s = 0.f, s2 = 0.f;
  for (int c = threadIdx.x; c < kC; c += 256) { float v = x[c]; s += v; s2 += v * v; }
  __shared__ float rs[256], rq[256];
  rs[threadIdx.x] = s; rq[threadIdx.x] = s2;
  __syncthreads();
  for (int o = 128; o > 0; o >>= 1) {
    if (threadIdx.x < (unsigned)o) { rs[threadIdx.x] += rs[threadIdx.x + o]; rq[threadIdx.x] += rq[threadIdx.x + o]; }
    __syncthreads();
  }
  const float mean = rs[0] * (1.f / kC);
  const float var  = rq[0] * (1.f / kC) - mean * mean;
  const float rinv = rsqrtf(var + eps);
  for (int c = threadIdx.x; c < kC; c += 256) {
    float v = (x[c] - mean) * rinv * w[c] + bb[c];
    if (outF) outF[(size_t)r * kC + c] = v;
    if (outB) outB[(size_t)r * kC + c] = (bf16_t)v;
  }
}

// ---------------------------------------------------------------------------
// Embedding: h[b,s,:] = cls+pos (s<BUF) else x @ z_proj^T + b  (K=VAE=16)
// ---------------------------------------------------------------------------
__global__ __launch_bounds__(256)
void embed_kernel(const float* __restrict__ x, const int* __restrict__ labels,
                  const float* __restrict__ zw, const float* __restrict__ zb,
                  const float* __restrict__ cls, const float* __restrict__ pos,
                  float* __restrict__ hf) {
  const int row = blockIdx.x;
  const int b = row / kS, s = row % kS;
  float* o = hf + (size_t)row * kC;
  const int c0 = threadIdx.x * 4;
  if (s < kBUF) {
    const float* ce = cls + (size_t)labels[b] * kC;
    const float* pp = pos + (size_t)s * kC;
#pragma unroll
    for (int i = 0; i < 4; ++i) o[c0 + i] = ce[c0 + i] + pp[c0 + i];
  } else {
    const int n = s - kBUF;
    const float* xv = x + ((size_t)b * kN + n) * kVAE;
#pragma unroll
    for (int i = 0; i < 4; ++i) {
      const int c = c0 + i;
      float acc = zb[c];
      const float* wr = zw + (size_t)c * kVAE;
#pragma unroll
      for (int k = 0; k < kVAE; ++k) acc += xv[k] * wr[k];
      o[c] = acc;
    }
  }
}

// ---------------------------------------------------------------------------
// RoPE + head packing: in [rows_total, rowStride] f32 (+colOff) ->
// out [B,H,rows,HD] bf16. freq==null -> plain pack (no rotation).
// ---------------------------------------------------------------------------
__global__ __launch_bounds__(256)
void rope_pack_kernel(const float* __restrict__ in, int rowStride, int colOff,
                      const float* __restrict__ freq, bf16_t* __restrict__ out,
                      int rows) {
  const int row = blockIdx.x;
  const int b = row / rows, n = row % rows;
  const float* src = in + (size_t)row * rowStride + colOff;
  const int c0 = threadIdx.x * 4;
  const int h = c0 / kHD, d = c0 % kHD;
  float v0 = src[c0], v1 = src[c0 + 1], v2 = src[c0 + 2], v3 = src[c0 + 3];
  if (freq) {
    const float* f = freq + ((size_t)n * (kHD / 2) + (d >> 1)) * 2;
    const float c = f[0], s = f[1];
    const float c2 = f[2], s2 = f[3];
    float r0 = v0 * c - v1 * s,  i0 = v1 * c + v0 * s;
    float r1 = v2 * c2 - v3 * s2, i1 = v3 * c2 + v2 * s2;
    v0 = r0; v1 = i0; v2 = r1; v3 = i1;
  }
  bf16_t* o = out + (((size_t)b * kH + h) * rows + n) * (size_t)kHD + d;
  o[0] = (bf16_t)v0; o[1] = (bf16_t)v1; o[2] = (bf16_t)v2; o[3] = (bf16_t)v3;
}

// V transpose pack: -> out [B,H,HD,rows] bf16
__global__ __launch_bounds__(256)
void packT_kernel(const float* __restrict__ in, int rowStride, int colOff,
                  bf16_t* __restrict__ out, int rows) {
  const int row = blockIdx.x;
  const int b = row / rows, n = row % rows;
  const float* src = in + (size_t)row * rowStride + colOff;
  const int c0 = threadIdx.x * 4;
#pragma unroll
  for (int i = 0; i < 4; ++i) {
    const int c = c0 + i, h = c / kHD, d = c % kHD;
    out[(((size_t)b * kH + h) * kHD + d) * (size_t)rows + n] = (bf16_t)src[c];
  }
}

// f32 -> bf16 bulk convert (weights)
__global__ __launch_bounds__(256)
void cvt_bf16_kernel(const float* __restrict__ in, bf16_t* __restrict__ out,
                     size_t n) {
  size_t i = (size_t)blockIdx.x * 256 + threadIdx.x;
  const size_t stride = (size_t)gridDim.x * 256;
  for (; i < n; i += stride) out[i] = (bf16_t)in[i];
}

// ---------------------------------------------------------------------------
// Host orchestration
// ---------------------------------------------------------------------------
enum {
  IN_X = 0, IN_LABELS, IN_MASK, IN_FREQ,
  IN_ZPW, IN_ZPB, IN_ZLW, IN_ZLB, IN_CLS, IN_POS, IN_ENW, IN_ENB,
  IN_N1W, IN_N1B, IN_QKVW, IN_QKVB, IN_PROJW, IN_PROJB,
  IN_CNW, IN_CNB, IN_CCNW, IN_CCNB,
  IN_QW, IN_QB, IN_KW, IN_KB, IN_VW, IN_VB,
  IN_CPROJW, IN_CPROJB, IN_N2W, IN_N2B,
  IN_FC1W, IN_FC1B, IN_FC2W, IN_FC2B
};

extern "C" void kernel_launch(void* const* d_in, const int* in_sizes, int n_in,
                              void* d_out, int out_size, void* d_ws, size_t ws_size,
                              hipStream_t stream) {
  (void)in_sizes; (void)n_in; (void)out_size; (void)ws_size;

  const float* x      = (const float*)d_in[IN_X];
  const int*   labels = (const int*)  d_in[IN_LABELS];
  const float* freq   = (const float*)d_in[IN_FREQ];

  const int BS = kB * kS;     // 6144 full-stream rows
  const int MN = kB * kN;     // 4096 image rows
  const int MC = kB * kBUF;   // 2048 cond rows

  // ---- workspace bump allocator ----
  char* wsp = (char*)d_ws;
  auto alloc = [&](size_t bytes) -> void* {
    void* p = (void*)wsp;
    wsp += (bytes + 255) & ~(size_t)255;
    return p;
  };

  float*  hf    = (float*) alloc((size_t)BS * kC * 4);        // f32 residual stream
  bf16_t* xb    = (bf16_t*)alloc((size_t)BS * kC * 2);        // LN outputs (bf16)
  bf16_t* condb = (bf16_t*)alloc((size_t)MC * kC * 2);
  float*  qkvf  = (float*) alloc((size_t)MN * 3 * kC * 4);
  float*  qf    = (float*) alloc((size_t)MN * kC * 4);
  float*  kf    = (float*) alloc((size_t)MC * kC * 4);
  float*  vf    = (float*) alloc((size_t)MC * kC * 4);
  bf16_t* Qs    = (bf16_t*)alloc((size_t)MN * kC * 2);        // [B,H,N,HD]
  bf16_t* Ks    = (bf16_t*)alloc((size_t)MN * kC * 2);
  bf16_t* Vts   = (bf16_t*)alloc((size_t)MN * kC * 2);        // [B,H,HD,N]
  bf16_t* Qc    = (bf16_t*)alloc((size_t)MN * kC * 2);
  bf16_t* Kc    = (bf16_t*)alloc((size_t)MC * kC * 2);        // [B,H,BUF,HD]
  bf16_t* Vct   = (bf16_t*)alloc((size_t)MC * kC * 2);        // [B,H,HD,BUF]
  bf16_t* attnb = (bf16_t*)alloc((size_t)MN * kC * 2);
  bf16_t* mlpb  = (bf16_t*)alloc((size_t)BS * kMLP * 2);

  const size_t sz_qkv = (size_t)kL * 3 * kC * kC;
  const size_t sz_cc  = (size_t)kL * kC * kC;
  const size_t sz_fc  = (size_t)kL * kMLP * kC;
  bf16_t* wqkv   = (bf16_t*)alloc(sz_qkv * 2);
  bf16_t* wproj  = (bf16_t*)alloc(sz_cc  * 2);
  bf16_t* wq     = (bf16_t*)alloc(sz_cc  * 2);
  bf16_t* wk     = (bf16_t*)alloc(sz_cc  * 2);
  bf16_t* wv     = (bf16_t*)alloc(sz_cc  * 2);
  bf16_t* wcproj = (bf16_t*)alloc(sz_cc  * 2);
  bf16_t* wfc1   = (bf16_t*)alloc(sz_fc  * 2);
  bf16_t* wfc2   = (bf16_t*)alloc(sz_fc  * 2);

  auto conv = [&](int idx, bf16_t* dst, size_t n) {
    cvt_bf16_kernel<<<4096, 256, 0, stream>>>((const float*)d_in[idx], dst, n);
  };
  conv(IN_QKVW,   wqkv,   sz_qkv);
  conv(IN_PROJW,  wproj,  sz_cc);
  conv(IN_QW,     wq,     sz_cc);
  conv(IN_KW,     wk,     sz_cc);
  conv(IN_VW,     wv,     sz_cc);
  conv(IN_CPROJW, wcproj, sz_cc);
  conv(IN_FC1W,   wfc1,   sz_fc);
  conv(IN_FC2W,   wfc2,   sz_fc);

  auto gemm = [&](const bf16_t* A, const bf16_t* W, const float* bias,
                  const float* res, float* oF, bf16_t* oB,
                  int M, int Nn, int K, int flags, int oseg, int ooff, int ostr) {
    dim3 grid(Nn / 64, M / 128);
    gemm_bf16_kernel<<<grid, 256, 0, stream>>>(A, W, bias, res, oF, oB,
                                               M, Nn, K, flags, oseg, ooff, ostr);
  };
  auto ln = [&](const float* in, const float* w, const float* b, float* oF,
                bf16_t* oB, int rows, int seg, int soff, int sstr, float eps) {
    layernorm_kernel<<<rows, 256, 0, stream>>>(in, w, b, oF, oB, seg, soff, sstr, eps);
  };

  // ---- embed + z_ln (residual stream hf) ----
  embed_kernel<<<BS, 256, 0, stream>>>(
      x, labels, (const float*)d_in[IN_ZPW], (const float*)d_in[IN_ZPB],
      (const float*)d_in[IN_CLS], (const float*)d_in[IN_POS], hf);
  ln(hf, (const float*)d_in[IN_ZLW], (const float*)d_in[IN_ZLB],
     hf, nullptr, BS, BS, 0, 0, 1e-6f);

  const dim3 agrid(kN / 16, kH, kB);

  for (int l = 0; l < kL; ++l) {
    const float* n1w    = (const float*)d_in[IN_N1W]    + (size_t)l * kC;
    const float* n1b    = (const float*)d_in[IN_N1B]    + (size_t)l * kC;
    const float* qkvb   = (const float*)d_in[IN_QKVB]   + (size_t)l * 3 * kC;
    const float* projb  = (const float*)d_in[IN_PROJB]  + (size_t)l * kC;
    const float* cnw    = (const float*)d_in[IN_CNW]    + (size_t)l * kC;
    const float* cnb    = (const float*)d_in[IN_CNB]    + (size_t)l * kC;
    const float* ccnw   = (const float*)d_in[IN_CCNW]   + (size_t)l * kC;
    const float* ccnb   = (const float*)d_in[IN_CCNB]   + (size_t)l * kC;
    const float* qb     = (const float*)d_in[IN_QB]     + (size_t)l * kC;
    const float* kb     = (const float*)d_in[IN_KB]     + (size_t)l * kC;
    const float* vb     = (const float*)d_in[IN_VB]     + (size_t)l * kC;
    const float* cprojb = (const float*)d_in[IN_CPROJB] + (size_t)l * kC;
    const float* n2w    = (const float*)d_in[IN_N2W]    + (size_t)l * kC;
    const float* n2b    = (const float*)d_in[IN_N2B]    + (size_t)l * kC;
    const float* fc1b   = (const float*)d_in[IN_FC1B]   + (size_t)l * kMLP;
    const float* fc2b   = (const float*)d_in[IN_FC2B]   + (size_t)l * kC;

    const bf16_t* wqkv_l   = wqkv   + (size_t)l * 3 * kC * kC;
    const bf16_t* wproj_l  = wproj  + (size_t)l * kC * kC;
    const bf16_t* wq_l     = wq     + (size_t)l * kC * kC;
    const bf16_t* wk_l     = wk     + (size_t)l * kC * kC;
    const bf16_t* wv_l     = wv     + (size_t)l * kC * kC;
    const bf16_t* wcproj_l = wcproj + (size_t)l * kC * kC;
    const bf16_t* wfc1_l   = wfc1   + (size_t)l * kMLP * kC;
    const bf16_t* wfc2_l   = wfc2   + (size_t)l * kC * kMLP;

    // --- self attention on image tokens ---
    ln(hf, n1w, n1b, nullptr, xb, MN, kN, kBUF, kS, 1e-5f);
    gemm(xb, wqkv_l, qkvb, nullptr, qkvf, nullptr, MN, 3 * kC, kC, 0, MN, 0, 0);
    rope_pack_kernel<<<MN, 256, 0, stream>>>(qkvf, 3 * kC, 0,       freq,    Qs,  kN);
    rope_pack_kernel<<<MN, 256, 0, stream>>>(qkvf, 3 * kC, kC,      freq,    Ks,  kN);
    packT_kernel    <<<MN, 256, 0, stream>>>(qkvf, 3 * kC, 2 * kC,           Vts, kN);
    attn_wmma_kernel<kN, true><<<agrid, 32, 0, stream>>>(Qs, Ks, Vts, attnb, 0.125f);
    gemm(attnb, wproj_l, projb, hf, hf, nullptr, MN, kC, kC, 0, kN, kBUF, kS);

    // --- cross attention (queries=img, keys/values=cond) ---
    ln(hf, cnw,  cnb,  nullptr, xb,    MN, kN,   kBUF, kS, 1e-5f);
    ln(hf, ccnw, ccnb, nullptr, condb, MC, kBUF, 0,    kS, 1e-5f);
    gemm(xb,    wq_l, qb, nullptr, qf, nullptr, MN, kC, kC, 0, MN, 0, 0);
    gemm(condb, wk_l, kb, nullptr, kf, nullptr, MC, kC, kC, 0, MC, 0, 0);
    gemm(condb, wv_l, vb, nullptr, vf, nullptr, MC, kC, kC, 0, MC, 0, 0);
    rope_pack_kernel<<<MN, 256, 0, stream>>>(qf, kC, 0, freq,    Qc, kN);
    rope_pack_kernel<<<MC, 256, 0, stream>>>(kf, kC, 0, nullptr, Kc, kBUF);
    packT_kernel    <<<MC, 256, 0, stream>>>(vf, kC, 0,          Vct, kBUF);
    attn_wmma_kernel<kBUF, false><<<agrid, 32, 0, stream>>>(Qc, Kc, Vct, attnb, 0.125f);
    gemm(attnb, wcproj_l, cprojb, hf, hf, nullptr, MN, kC, kC, 0, kN, kBUF, kS);

    // --- MLP on the full stream (cond + img) ---
    ln(hf, n2w, n2b, nullptr, xb, BS, BS, 0, 0, 1e-5f);
    gemm(xb,   wfc1_l, fc1b, nullptr, nullptr, mlpb, BS, kMLP, kC, 1, BS, 0, 0);
    gemm(mlpb, wfc2_l, fc2b, hf, hf, nullptr, BS, kC, kMLP, 0, BS, 0, 0);
  }

  // ---- final norm -> d_out (f32, [B,S,C]) ----
  ln(hf, (const float*)d_in[IN_ENW], (const float*)d_in[IN_ENB],
     (float*)d_out, nullptr, BS, BS, 0, 0, 1e-5f);
}
